// NodeProcessor_78915729097029
// MI455X (gfx1250) — matrix operand order repaired and checked by reading, not compiled, not run on hardware
//
#include <hip/hip_runtime.h>
#include <hip/hip_bf16.h>

typedef __bf16 v16bf __attribute__((ext_vector_type(16)));
typedef float  v8f   __attribute__((ext_vector_type(8)));

#define S1C   64
#define V1C   32
#define S2C   16
#define V2C   8
#define SOUT  96
#define NNODES 10000
#define NEDGES 160000
#define KDIM  1280
#define KCH   40      // 1280 / 32
#define NTS   6       // 96 / 16
#define NTV   2       // 32 / 16

// workspace byte offsets
#define WS_WSB   0                            // bf16 [1280*96]   = 245760 B
#define WS_WVB   245760                       // bf16 [1280*32]   =  81920 B
#define WS_SN    327680                       // f32  [10000*64]  = 2560000 B
#define WS_VN    (327680 + 2560000)           // f32  [10000*96]  = 3840000 B
#define WS_STATS (327680 + 2560000 + 3840000) // f32  [160]

// ---------------------------------------------------------------------------
// Kernel 0: convert + scale + swizzle weights into WMMA B-fragment layout.
// Fragment (kc, nt) = 512 bf16; lane L owns 16 contiguous bf16 (32 B):
//   n = nt*16 + (L&15),  k = kc*32 + e + 8*(e>>3) + 8*(L>>4)   (e = 0..15)
// ---------------------------------------------------------------------------
__global__ void wconv_kernel(const float* __restrict__ Wss,
                             const float* __restrict__ Wvvs,
                             const float* __restrict__ Wsv,
                             const float* __restrict__ Wvs,
                             const float* __restrict__ Wvvv,
                             __bf16* __restrict__ WsB,
                             __bf16* __restrict__ WvB) {
  const float INV   = 0.02795084971874737f;   // 1/sqrt(1280)
  const float ISQ3  = 0.57735026918962576f;
  const float ISQ2  = 0.70710678118654752f;
  int id = blockIdx.x * blockDim.x + threadIdx.x;
  if (id < KDIM * SOUT) {
    int f = id >> 9, r = id & 511;
    int L = r >> 4, e = r & 15;
    int kc = f / NTS, nt = f - kc * NTS;
    int k = kc * 32 + e + 8 * (e >> 3) + 8 * (L >> 4);
    int n = nt * 16 + (L & 15);
    float val;
    if (k < 1024) { int a = k >> 4, b = k & 15;
      val = Wss[(a * 16 + b) * SOUT + n] * INV;
    } else { int k2 = k - 1024; int a = k2 >> 3, b = k2 & 7;
      val = Wvvs[(a * 8 + b) * SOUT + n] * (INV * ISQ3);
    }
    WsB[id] = (__bf16)val;
  } else {
    int id2 = id - KDIM * SOUT;
    if (id2 >= KDIM * V1C) return;
    int f = id2 >> 9, r = id2 & 511;
    int L = r >> 4, e = r & 15;
    int kc = f / NTV, nt = f - kc * NTV;
    int k = kc * 32 + e + 8 * (e >> 3) + 8 * (L >> 4);
    int n = nt * 16 + (L & 15);
    float val;
    if (k < 512) { int a = k >> 3, b = k & 7;
      val = Wsv[(a * 8 + b) * V1C + n] * INV;
    } else if (k < 1024) { int k2 = k - 512; int a = k2 >> 4, b = k2 & 15;
      val = Wvs[(a * 16 + b) * V1C + n] * INV;
    } else { int k2 = k - 1024; int a = k2 >> 3, b = k2 & 7;
      val = Wvvv[(a * 8 + b) * V1C + n] * (INV * ISQ2);
    }
    WvB[id2] = (__bf16)val;
  }
}

// ---------------------------------------------------------------------------
// Kernel 1: edge kernel. One wave32 per 16-edge tile; 4 waves per block.
// A-fragment lane mapping follows the documented 16-bit A 16x32 layout:
//   row M = lane&15, K_local = e + 8*(e>>3) + 8*(lane>>4)
// ---------------------------------------------------------------------------
struct Tile {
  float s1[16][S1C];
  float v1[16][V1C][3];
  float s2[16][S2C];
  float v2[16][V2C][3];
};

__global__ __launch_bounds__(128)
void edge_kernel(const float* __restrict__ x,
                 const float* __restrict__ ea,
                 const long long* __restrict__ ei,
                 const __bf16* __restrict__ WsB,
                 const __bf16* __restrict__ WvB,
                 float* __restrict__ s_n,
                 float* __restrict__ v_n) {
  __shared__ Tile tiles[4];
  const int lane = threadIdx.x & 31;
  const int wave = threadIdx.x >> 5;
  const int tile = blockIdx.x * 4 + wave;
  const int t0   = tile * 16;
  Tile& T = tiles[wave];

  // gather node features x[i[e]] and edge attrs into LDS
  for (int idx = lane; idx < 16 * 160; idx += 32) {
    int e = idx / 160, f = idx - e * 160;
    int node = (int)ei[t0 + e];
    float v = x[(size_t)node * 160 + f];
    if (f < S1C) T.s1[e][f] = v;
    else { int g = f - S1C; T.v1[e][g / 3][g % 3] = v; }
  }
  for (int idx = lane; idx < 16 * 40; idx += 32) {
    int e = idx / 40, f = idx - e * 40;
    float v = ea[(size_t)t0 * 40 + idx];
    if (f < S2C) T.s2[e][f] = v;
    else { int g = f - S2C; T.v2[e][g / 3][g % 3] = v; }
  }
  __syncthreads();

  const int m    = lane & 15;
  const int half = lane >> 4;

  int nd[8];
  #pragma unroll
  for (int r = 0; r < 8; ++r) nd[r] = (int)ei[t0 + r + 8 * half];

  // ---- scalar-channel GEMM: [16 x 1280] x [1280 x 96], bf16 -> f32 ----
  v8f accS[NTS] = {};
  for (int kc = 0; kc < KCH; ++kc) {
    // issue all 6 B-fragment loads first so they overlap the A build
    const v16bf* Bp = (const v16bf*)(WsB + (size_t)kc * NTS * 512);
    v16bf B[NTS];
    #pragma unroll
    for (int t = 0; t < NTS; ++t) B[t] = Bp[t * 32 + lane];

    // prefetch next chunk's B fragments (6144 B, spread across lanes)
    {
      int kn = (kc + 1) % KCH;
      const char* nxt = (const char*)WsB + (size_t)kn * NTS * 1024 + lane * 192;
      __builtin_prefetch(nxt, 0, 0);
    }

    v16bf A;
    if (kc < 32) {            // s1 (x) s2 outer-product block
      #pragma unroll
      for (int e = 0; e < 16; ++e) {
        int kk = kc * 32 + e + 8 * (e >> 3) + 8 * half;
        int a = kk >> 4, b = kk & 15;
        A[e] = (__bf16)(T.s1[m][a] * T.s2[m][b]);
      }
    } else {                  // v1 . v2 dot-product block
      #pragma unroll
      for (int e = 0; e < 16; ++e) {
        int kk = kc * 32 + e + 8 * (e >> 3) + 8 * half - 1024;
        int a = kk >> 3, b = kk & 7;
        float d = T.v1[m][a][0] * T.v2[m][b][0]
                + T.v1[m][a][1] * T.v2[m][b][1]
                + T.v1[m][a][2] * T.v2[m][b][2];
        A[e] = (__bf16)d;
      }
    }
    #pragma unroll
    for (int t = 0; t < NTS; ++t)
      accS[t] = __builtin_amdgcn_wmma_f32_16x16x32_bf16(
          false, A, false, B[t], (short)0, accS[t], false, false);
  }

  // SiLU on feature channels (cols 0..63) -> scatter-add into node scalars
  #pragma unroll
  for (int t = 0; t < 4; ++t) {
    int col = t * 16 + m;
    #pragma unroll
    for (int r = 0; r < 8; ++r) {
      float v = accS[t][r];
      float f = v / (1.f + __expf(-v));
      atomicAdd(&s_n[(size_t)nd[r] * S1C + col], f);
    }
  }
  // sigmoid gates (cols 64..95), one gate per vector channel
  float gate[2][8];
  #pragma unroll
  for (int t = 0; t < 2; ++t)
    #pragma unroll
    for (int r = 0; r < 8; ++r)
      gate[t][r] = 1.f / (1.f + __expf(-accS[4 + t][r]));

  // ---- vector-channel GEMM: [16 x 1280] x [1280 x 32] for p = 0,1,2 ----
  // kc outer / p inner: each B fragment is loaded ONCE and reused by all
  // three spatial components; v1/v2 LDS reads are shared across components.
  v8f accV[3][NTV] = {};
  for (int kc = 0; kc < KCH; ++kc) {
    const v16bf* Bp = (const v16bf*)(WvB + (size_t)kc * NTV * 512);
    v16bf B0 = Bp[lane];
    v16bf B1 = Bp[32 + lane];

    {
      int kn = (kc + 1) % KCH;
      const char* nxt = (const char*)WvB + (size_t)kn * NTV * 1024 + lane * 64;
      __builtin_prefetch(nxt, 0, 0);
    }

    v16bf Ap[3];
    if (kc < 16) {            // s1 (x) v2_p
      #pragma unroll
      for (int e = 0; e < 16; ++e) {
        int kk = kc * 32 + e + 8 * (e >> 3) + 8 * half;
        int a = kk >> 3, b = kk & 7;
        float sa = T.s1[m][a];
        float bx = T.v2[m][b][0], by = T.v2[m][b][1], bz = T.v2[m][b][2];
        Ap[0][e] = (__bf16)(sa * bx);
        Ap[1][e] = (__bf16)(sa * by);
        Ap[2][e] = (__bf16)(sa * bz);
      }
    } else if (kc < 32) {     // v1_p (x) s2
      #pragma unroll
      for (int e = 0; e < 16; ++e) {
        int kk = kc * 32 + e + 8 * (e >> 3) + 8 * half - 512;
        int a = kk >> 4, b = kk & 15;
        float sb = T.s2[m][b];
        float ax = T.v1[m][a][0], ay = T.v1[m][a][1], az = T.v1[m][a][2];
        Ap[0][e] = (__bf16)(ax * sb);
        Ap[1][e] = (__bf16)(ay * sb);
        Ap[2][e] = (__bf16)(az * sb);
      }
    } else {                  // (v1 x v2)_p  (Levi-Civita cross products)
      #pragma unroll
      for (int e = 0; e < 16; ++e) {
        int kk = kc * 32 + e + 8 * (e >> 3) + 8 * half - 1024;
        int a = kk >> 3, b = kk & 7;
        float ax = T.v1[m][a][0], ay = T.v1[m][a][1], az = T.v1[m][a][2];
        float bx = T.v2[m][b][0], by = T.v2[m][b][1], bz = T.v2[m][b][2];
        Ap[0][e] = (__bf16)(ay * bz - az * by);
        Ap[1][e] = (__bf16)(az * bx - ax * bz);
        Ap[2][e] = (__bf16)(ax * by - ay * bx);
      }
    }
    #pragma unroll
    for (int p = 0; p < 3; ++p) {
      accV[p][0] = __builtin_amdgcn_wmma_f32_16x16x32_bf16(
          false, Ap[p], false, B0, (short)0, accV[p][0], false, false);
      accV[p][1] = __builtin_amdgcn_wmma_f32_16x16x32_bf16(
          false, Ap[p], false, B1, (short)0, accV[p][1], false, false);
    }
  }

  #pragma unroll
  for (int p = 0; p < 3; ++p)
    #pragma unroll
    for (int t = 0; t < NTV; ++t) {
      int c = t * 16 + m;
      #pragma unroll
      for (int r = 0; r < 8; ++r) {
        float v = accV[p][t][r] * gate[t][r];
        atomicAdd(&v_n[(size_t)nd[r] * 96 + c * 3 + p], v);
      }
    }
}

// ---------------------------------------------------------------------------
// Kernel 2: per-channel statistics over nodes.
// blocks 0..63  : mean + rstd of aggregated scalars
// blocks 64..95 : rstd of vector norms
// ---------------------------------------------------------------------------
__global__ void stats_kernel(const float* __restrict__ s_n,
                             const float* __restrict__ v_n,
                             float* __restrict__ stats) {
  __shared__ float rs[256], rq[256];
  const int c = blockIdx.x;
  const int tid = threadIdx.x;
  if (c < S1C) {
    float s = 0.f, q = 0.f;
    for (int n = tid; n < NNODES; n += 256) {
      float v = s_n[(size_t)n * S1C + c];
      s += v; q += v * v;
    }
    rs[tid] = s; rq[tid] = q;
    __syncthreads();
    for (int w = 128; w > 0; w >>= 1) {
      if (tid < w) { rs[tid] += rs[tid + w]; rq[tid] += rq[tid + w]; }
      __syncthreads();
    }
    if (tid == 0) {
      float mu  = rs[0] / (float)NNODES;
      float var = rq[0] / (float)NNODES - mu * mu;
      stats[c]       = mu;
      stats[64 + c]  = rsqrtf(var + 1e-5f);
    }
  } else {
    const int cc = c - S1C;
    float s = 0.f;
    for (int n = tid; n < NNODES; n += 256) {
      const float* vp = &v_n[(size_t)n * 96 + cc * 3];
      s += vp[0] * vp[0] + vp[1] * vp[1] + vp[2] * vp[2];
    }
    rs[tid] = s;
    __syncthreads();
    for (int w = 128; w > 0; w >>= 1) {
      if (tid < w) rs[tid] += rs[tid + w];
      __syncthreads();
    }
    if (tid == 0) {
      float vnorm = rs[0] / (3.f * (float)NNODES);
      stats[128 + cc] = rsqrtf(vnorm + 1e-5f);
    }
  }
}

// ---------------------------------------------------------------------------
// Kernel 3: normalize + residual add -> output
// ---------------------------------------------------------------------------
__global__ void finalize_kernel(const float* __restrict__ x,
                                const float* __restrict__ s_n,
                                const float* __restrict__ v_n,
                                const float* __restrict__ stats,
                                const float* __restrict__ bn_w_s,
                                const float* __restrict__ bn_b_s,
                                const float* __restrict__ bn_w_v,
                                float* __restrict__ out) {
  int t = blockIdx.x * blockDim.x + threadIdx.x;
  if (t >= NNODES * 160) return;
  int n = t / 160, f = t - n * 160;
  float o;
  if (f < S1C) {
    o = (s_n[(size_t)n * S1C + f] - stats[f]) * stats[64 + f] * bn_w_s[f] + bn_b_s[f];
  } else {
    int g = f - S1C;
    int c = g / 3;
    o = v_n[(size_t)n * 96 + g] * stats[128 + c] * bn_w_v[c];
  }
  out[t] = o + x[t];
}

// ---------------------------------------------------------------------------
extern "C" void kernel_launch(void* const* d_in, const int* in_sizes, int n_in,
                              void* d_out, int out_size, void* d_ws, size_t ws_size,
                              hipStream_t stream) {
  const float*     x    = (const float*)d_in[0];
  const float*     ea   = (const float*)d_in[1];
  const float*     Wss  = (const float*)d_in[2];
  const float*     Wvvs = (const float*)d_in[3];
  const float*     Wsv  = (const float*)d_in[4];
  const float*     Wvs  = (const float*)d_in[5];
  const float*     Wvvv = (const float*)d_in[6];
  const float*     bnws = (const float*)d_in[7];
  const float*     bnbs = (const float*)d_in[8];
  const float*     bnwv = (const float*)d_in[9];
  const long long* ei   = (const long long*)d_in[10];

  char* ws = (char*)d_ws;
  __bf16* WsB   = (__bf16*)(ws + WS_WSB);
  __bf16* WvB   = (__bf16*)(ws + WS_WVB);
  float*  s_n   = (float*)(ws + WS_SN);
  float*  v_n   = (float*)(ws + WS_VN);
  float*  stats = (float*)(ws + WS_STATS);

  // zero the node accumulators (graph-capturable async memset)
  hipMemsetAsync(s_n, 0, 2560000 + 3840000, stream);

  // weights -> scaled bf16 WMMA fragments
  wconv_kernel<<<640, 256, 0, stream>>>(Wss, Wvvs, Wsv, Wvs, Wvvv, WsB, WvB);

  // per-edge WMMA GEMMs + gating + scatter-add (10000 tiles, 4 waves/block)
  edge_kernel<<<2500, 128, 0, stream>>>(x, ea, ei, WsB, WvB, s_n, v_n);

  // channel statistics
  stats_kernel<<<96, 256, 0, stream>>>(s_n, v_n, stats);

  // batch-norm + residual
  finalize_kernel<<<6250, 256, 0, stream>>>(x, s_n, v_n, stats, bnws, bnbs, bnwv,
                                            (float*)d_out);
}
